// RAN_44100724195596
// MI455X (gfx1250) — matrix operand order repaired and checked by reading, not compiled, not run on hardware
//
#include <hip/hip_runtime.h>
#include <stdint.h>

// RAN cell forward for MI455X (gfx1250, wave32, WMMA).
// Phase 0: convert input+weights f32->bf16 (v_cvt_pk_bf16_f32), init h + barrier.
// Phase 1: x_c/x_i/x_f = input @ W^T + b  (bf16 WMMA, f32 accum, LDS-staged B,
//          GLOBAL_LOAD_ASYNC_TO_LDS_B128 + s_wait_asynccnt double buffering)
// Phase 2: persistent scan kernel, 32 WGs, 1 grid barrier per time step.

#define T_DIM 512
#define B_DIM 64
#define I_DIM 512
#define H_DIM 1024
#define SCAN_WGS 32
#define BS_STRIDE 40   // LDS column stride in bf16 elems (80B: 16B-aligned, bank-spread)

typedef __attribute__((ext_vector_type(16))) __bf16 bf16x16;
typedef __attribute__((ext_vector_type(8)))  __bf16 bf16x8;
typedef __attribute__((ext_vector_type(8)))  float  f32x8;
typedef int v4i __attribute__((vector_size(16)));   // matches builtin's V4i param

// ---------- async global->LDS availability probe ----------
#if defined(__has_builtin)
#if __has_builtin(__builtin_amdgcn_global_load_async_to_lds_b128)
#define HAVE_ASYNC_LDS 1
#endif
#endif
#ifndef HAVE_ASYNC_LDS
#define HAVE_ASYNC_LDS 0
#endif

#if HAVE_ASYNC_LDS
__device__ __forceinline__ void async_wait0() {
#if __has_builtin(__builtin_amdgcn_s_wait_asynccnt)
  __builtin_amdgcn_s_wait_asynccnt(0);
#else
  asm volatile("s_wait_asynccnt 0x0" ::: "memory");
#endif
}
#endif

// ---------- scalar helpers (hardware bf16 conversion) ----------
__device__ __forceinline__ uint16_t f2bf(float f) {
  return __builtin_bit_cast(uint16_t, (__bf16)f);
}
__device__ __forceinline__ float bf2f(uint16_t u) {
  union { uint32_t u; float f; } x; x.u = ((uint32_t)u) << 16; return x.f;
}
__device__ __forceinline__ float sigm(float x) {
  return 1.0f / (1.0f + __expf(-x));
}
__device__ __forceinline__ f32x8 zero8() {
  f32x8 z = {0.f,0.f,0.f,0.f,0.f,0.f,0.f,0.f}; return z;
}

// ---------- WMMA fragment loads (16x16x32 bf16, contiguous-K order shared by A and B) ----------
// A (16x32 MxK): row = lane&15, 16 contiguous K at k0+16*half (half=lane>>4)
__device__ __forceinline__ bf16x16 load_a_bf(const uint16_t* row, int k0, int half) {
  const uint16_t* p0 = row + k0 + 16 * half;
  bf16x8 lo = *reinterpret_cast<const bf16x8*>(p0);
  bf16x8 hi = *reinterpret_cast<const bf16x8*>(p0 + 8);
  return __builtin_shufflevector(lo, hi, 0,1,2,3,4,5,6,7,8,9,10,11,12,13,14,15);
}
// B (32x16 KxN): lane holds col=lane&15, 16 contiguous K at k0+16*half.
__device__ __forceinline__ bf16x16 load_b_global(const uint16_t* colbase_khalf, int k0) {
  return *reinterpret_cast<const bf16x16*>(colbase_khalf + k0);
}
__device__ __forceinline__ bf16x16 load_b_lds(const uint16_t* p) {
  bf16x8 lo = *reinterpret_cast<const bf16x8*>(p);
  bf16x8 hi = *reinterpret_cast<const bf16x8*>(p + 8);
  return __builtin_shufflevector(lo, hi, 0,1,2,3,4,5,6,7,8,9,10,11,12,13,14,15);
}

__device__ __forceinline__ f32x8 wmma_bf16(bf16x16 a, bf16x16 b, f32x8 c) {
  return __builtin_amdgcn_wmma_f32_16x16x32_bf16(false, a, false, b, (short)0, c,
                                                 false, false);
}

// ---------- device-wide barrier (generation based, agent scope) ----------
__device__ __forceinline__ void grid_barrier(unsigned* bar, unsigned nwg) {
  __threadfence();          // release: publish this wave's stores
  __syncthreads();
  if (threadIdx.x == 0) {
    unsigned* cnt = bar;
    unsigned* gen = bar + 1;
    unsigned g = __hip_atomic_load(gen, __ATOMIC_RELAXED, __HIP_MEMORY_SCOPE_AGENT);
    unsigned a = __hip_atomic_fetch_add(cnt, 1u, __ATOMIC_ACQ_REL, __HIP_MEMORY_SCOPE_AGENT);
    if (a == nwg - 1u) {
      __hip_atomic_store(cnt, 0u, __ATOMIC_RELAXED, __HIP_MEMORY_SCOPE_AGENT);
      __hip_atomic_fetch_add(gen, 1u, __ATOMIC_RELEASE, __HIP_MEMORY_SCOPE_AGENT);
    } else {
      while (__hip_atomic_load(gen, __ATOMIC_ACQUIRE, __HIP_MEMORY_SCOPE_AGENT) == g) {
        __builtin_amdgcn_s_sleep(1);
      }
    }
  }
  __syncthreads();
  __threadfence();          // acquire: drop stale near-cache lines per wave
}

// ---------- phase 0 kernels ----------
__global__ void cvt_f32_bf16_kernel(const float* __restrict__ src,
                                    uint16_t* __restrict__ dst, int n) {
  int i = blockIdx.x * blockDim.x + threadIdx.x;
  int stride = gridDim.x * blockDim.x;
  for (; i < n; i += stride) dst[i] = f2bf(src[i]);
}

__global__ void init_kernel(const float* __restrict__ hidden,
                            float* __restrict__ hf, uint16_t* __restrict__ hb,
                            unsigned* __restrict__ bar) {
  int i = blockIdx.x * blockDim.x + threadIdx.x;
  if (i == 0) { bar[0] = 0u; bar[1] = 0u; }
  if (i < B_DIM * H_DIM) {
    float v = hidden[i];
    hf[i] = v;
    hb[i] = f2bf(v);
  }
}

// ---------- phase 1: input projections ----------
// grid (M/128, N/64, 3), block 256 (8 waves). Wave w computes rows
// m0=(bx*8+w)*16, cols n0..n0+63 (4 WMMA tiles, shared A fragment).
// B tile (64 cols x 32 K) staged once per block into LDS, double-buffered;
// one 16B async (or load+store) copy per thread per stage.
__device__ __forceinline__ void stageB(const uint16_t* __restrict__ W, int n0,
                                       int k0, uint16_t* __restrict__ dst,
                                       int tid) {
  const int c = tid >> 2;        // 0..63 column
  const int q = tid & 3;         // 0..3  16B chunk within 64B row
  const uint16_t* g = W + (size_t)(n0 + c) * I_DIM + k0 + q * 8;
  uint16_t* l = dst + c * BS_STRIDE + q * 8;
#if HAVE_ASYNC_LDS
  __builtin_amdgcn_global_load_async_to_lds_b128(
      (__attribute__((address_space(1))) v4i*)(uintptr_t)g,
      (__attribute__((address_space(3))) v4i*)l,
      0, 0);
#else
  *reinterpret_cast<int4*>(l) = *reinterpret_cast<const int4*>(g);
#endif
}

__global__ __launch_bounds__(256) void ran_proj_kernel(
    const uint16_t* __restrict__ in_bf,        // [T*B, I] bf16
    const uint16_t* __restrict__ wcx, const uint16_t* __restrict__ wix,
    const uint16_t* __restrict__ wfx,
    const float* __restrict__ bcx, const float* __restrict__ bix,
    const float* __restrict__ bfx,
    float* __restrict__ xc, uint16_t* __restrict__ xi, uint16_t* __restrict__ xf) {
  __shared__ uint16_t Bs[2][64 * BS_STRIDE];

  const int tid  = threadIdx.x;
  const int lane = tid & 31;
  const int wave = tid >> 5;
  const int half = lane >> 4;
  const int col  = lane & 15;
  const int m0 = (blockIdx.x * 8 + wave) * 16;
  const int n0 = blockIdx.y * 64;
  const int z  = blockIdx.z;
  const uint16_t* W  = (z == 0) ? wcx : (z == 1) ? wix : wfx;
  const float* bias  = (z == 0) ? bcx : (z == 1) ? bix : bfx;

  const uint16_t* Arow = in_bf + (size_t)(m0 + col) * I_DIM;

  stageB(W, n0, 0, Bs[0], tid);   // prologue

  f32x8 acc[4] = {zero8(), zero8(), zero8(), zero8()};
  for (int k0 = 0; k0 < I_DIM; k0 += 32) {
    const int buf = (k0 >> 5) & 1;
#if HAVE_ASYNC_LDS
    async_wait0();                // my wave's async copies into Bs[buf] landed
#endif
    __syncthreads();              // all waves' copies landed; Bs[buf^1] reusable
    if (k0 + 32 < I_DIM) stageB(W, n0, k0 + 32, Bs[buf ^ 1], tid);

    bf16x16 a = load_a_bf(Arow, k0, half);
    bf16x16 bfrag[4];
#pragma unroll
    for (int j = 0; j < 4; ++j)
      bfrag[j] = load_b_lds(&Bs[buf][(16 * j + col) * BS_STRIDE + 16 * half]);
#pragma unroll
    for (int j = 0; j < 4; ++j)
      acc[j] = wmma_bf16(a, bfrag[j], acc[j]);
  }

  // store (mode branch hoisted out of the element loops)
  if (z == 0) {
#pragma unroll
    for (int j = 0; j < 4; ++j) {
      const int n = n0 + 16 * j + col;
      const float bv = bias[n];
#pragma unroll
      for (int v = 0; v < 8; ++v) {
        const int m = m0 + v + 8 * half;
        xc[(size_t)m * H_DIM + n] = acc[j][v] + bv;
      }
    }
  } else {
    uint16_t* __restrict__ dst = (z == 1) ? xi : xf;
#pragma unroll
    for (int j = 0; j < 4; ++j) {
      const int n = n0 + 16 * j + col;
      const float bv = bias[n];
#pragma unroll
      for (int v = 0; v < 8; ++v) {
        const int m = m0 + v + 8 * half;
        dst[(size_t)m * H_DIM + n] = f2bf(acc[j][v] + bv);
      }
    }
  }
}

// ---------- phase 2: persistent recurrent scan ----------
// 32 WGs x 8 waves = 256 waves; wave gw owns tile (m0=(gw&3)*16 batch rows,
// n0=(gw>>2)*16 hidden cols) of BOTH gate GEMMs. One grid barrier per step.
__global__ __launch_bounds__(256) void ran_scan_kernel(
    const uint16_t* __restrict__ wic, const uint16_t* __restrict__ wfc,
    const float* __restrict__ bic, const float* __restrict__ bfc,
    const float* __restrict__ xc, const uint16_t* __restrict__ xi,
    const uint16_t* __restrict__ xf,
    float* __restrict__ hf,       // [2][B*H] f32
    uint16_t* __restrict__ hb,    // [2][B*H] bf16
    unsigned* __restrict__ bar,
    float* __restrict__ out) {    // [T*B*H] output ++ [B*H] hT
  const int lane = threadIdx.x & 31;
  const int half = lane >> 4;
  const int col  = lane & 15;
  const int gw = blockIdx.x * 8 + (threadIdx.x >> 5);   // 0..255
  const int m0 = (gw & 3) * 16;
  const int n0 = (gw >> 2) * 16;
  const int n  = n0 + col;

  const float bi_s = bic[n];
  const float bf_s = bfc[n];
  const uint16_t* Bi0 = wic + (size_t)n * H_DIM + 16 * half;
  const uint16_t* Bf0 = wfc + (size_t)n * H_DIM + 16 * half;

  for (int t = 0; t < T_DIM; ++t) {
    const int cur = t & 1, nxt = cur ^ 1;
    const uint16_t* Arow =
        hb + (size_t)cur * (B_DIM * H_DIM) + (size_t)(m0 + col) * H_DIM;

    const size_t xoff = (size_t)t * (B_DIM * H_DIM);
    const size_t idx0 = (size_t)(m0 + 8 * half) * H_DIM + n;
    __builtin_prefetch(&xc[xoff + idx0], 0, 1);   // warm step-t elementwise data
    __builtin_prefetch(&xi[xoff + idx0], 0, 1);
    __builtin_prefetch(&xf[xoff + idx0], 0, 1);

    f32x8 ai = zero8();
    f32x8 af = zero8();
    for (int k0 = 0; k0 < H_DIM; k0 += 32) {
      if (k0 + 32 < H_DIM) {                 // warm next tiles (global_prefetch)
        __builtin_prefetch(Bi0 + k0 + 32, 0, 1);
        __builtin_prefetch(Bf0 + k0 + 32, 0, 1);
      }
      bf16x16 a  = load_a_bf(Arow, k0, half);
      bf16x16 bi = load_b_global(Bi0, k0);
      bf16x16 bf = load_b_global(Bf0, k0);
      ai = wmma_bf16(a, bi, ai);
      af = wmma_bf16(a, bf, af);
    }

    const float*    hfc = hf + (size_t)cur * (B_DIM * H_DIM);
    float*          hfn = hf + (size_t)nxt * (B_DIM * H_DIM);
    uint16_t*       hbn = hb + (size_t)nxt * (B_DIM * H_DIM);
    float hnv[8];
#pragma unroll
    for (int v = 0; v < 8; ++v) {
      const int m = m0 + v + 8 * half;
      const size_t idx = (size_t)m * H_DIM + n;
      const float gi = sigm(ai[v] + bi_s + bf2f(xi[xoff + idx]));
      const float gf = sigm(af[v] + bf_s + bf2f(xf[xoff + idx]));
      const float c  = gi * xc[xoff + idx] + gf * hfc[idx];
      const float hn = tanhf(c);
      out[xoff + idx] = hn;
      hfn[idx] = hn;
      hbn[idx] = f2bf(hn);
      hnv[v] = hn;
    }
    if (t == T_DIM - 1) {                      // hT tail, hoisted out of hot loop
#pragma unroll
      for (int v = 0; v < 8; ++v) {
        const int m = m0 + v + 8 * half;
        out[(size_t)T_DIM * B_DIM * H_DIM + (size_t)m * H_DIM + n] = hnv[v];
      }
    }
    grid_barrier(bar, SCAN_WGS);   // h[nxt] fully published before next GEMM
  }
}

// ---------- workspace layout (bytes, all 256B aligned) ----------
static constexpr size_t OFF_BAR  = 0;                        // 256 B
static constexpr size_t OFF_WCX  = 256;                      // 1,048,576
static constexpr size_t OFF_WIX  = OFF_WCX  + 1048576;
static constexpr size_t OFF_WFX  = OFF_WIX  + 1048576;
static constexpr size_t OFF_WIC  = OFF_WFX  + 1048576;       // 2,097,152
static constexpr size_t OFF_WFC  = OFF_WIC  + 2097152;
static constexpr size_t OFF_INBF = OFF_WFC  + 2097152;       // 33,554,432 (bf16 input)
static constexpr size_t OFF_XC   = OFF_INBF + 33554432;      // 134,217,728 (f32)
static constexpr size_t OFF_XI   = OFF_XC   + 134217728;     // 67,108,864 (bf16)
static constexpr size_t OFF_XF   = OFF_XI   + 67108864;
static constexpr size_t OFF_HF   = OFF_XF   + 67108864;      // 2*B*H f32
static constexpr size_t OFF_HB   = OFF_HF   + 524288;        // 2*B*H bf16
// total ~310.1 MB

extern "C" void kernel_launch(void* const* d_in, const int* in_sizes, int n_in,
                              void* d_out, int out_size, void* d_ws, size_t ws_size,
                              hipStream_t stream) {
  const float* input  = (const float*)d_in[0];
  const float* hidden = (const float*)d_in[1];
  const float* w_cx   = (const float*)d_in[2];
  const float* w_ic   = (const float*)d_in[3];
  const float* w_ix   = (const float*)d_in[4];
  const float* w_fc   = (const float*)d_in[5];
  const float* w_fx   = (const float*)d_in[6];
  const float* b_cx   = (const float*)d_in[7];
  const float* b_ic   = (const float*)d_in[8];
  const float* b_ix   = (const float*)d_in[9];
  const float* b_fc   = (const float*)d_in[10];
  const float* b_fx   = (const float*)d_in[11];

  char* ws = (char*)d_ws;
  unsigned* bar    = (unsigned*)(ws + OFF_BAR);
  uint16_t* wcx_bf = (uint16_t*)(ws + OFF_WCX);
  uint16_t* wix_bf = (uint16_t*)(ws + OFF_WIX);
  uint16_t* wfx_bf = (uint16_t*)(ws + OFF_WFX);
  uint16_t* wic_bf = (uint16_t*)(ws + OFF_WIC);
  uint16_t* wfc_bf = (uint16_t*)(ws + OFF_WFC);
  uint16_t* in_bf  = (uint16_t*)(ws + OFF_INBF);
  float*    xc     = (float*)   (ws + OFF_XC);
  uint16_t* xi     = (uint16_t*)(ws + OFF_XI);
  uint16_t* xf     = (uint16_t*)(ws + OFF_XF);
  float*    hf     = (float*)   (ws + OFF_HF);
  uint16_t* hb     = (uint16_t*)(ws + OFF_HB);
  float*    out    = (float*)d_out;

  // Phase 0: bf16 copies (input once: removes 48x re-conversion from phase 1),
  // h init + barrier reset (every call, graph-safe).
  cvt_f32_bf16_kernel<<<2048, 256, 0, stream>>>(input, in_bf, T_DIM * B_DIM * I_DIM);
  cvt_f32_bf16_kernel<<<512, 256, 0, stream>>>(w_cx, wcx_bf, H_DIM * I_DIM);
  cvt_f32_bf16_kernel<<<512, 256, 0, stream>>>(w_ix, wix_bf, H_DIM * I_DIM);
  cvt_f32_bf16_kernel<<<512, 256, 0, stream>>>(w_fx, wfx_bf, H_DIM * I_DIM);
  cvt_f32_bf16_kernel<<<512, 256, 0, stream>>>(w_ic, wic_bf, H_DIM * H_DIM);
  cvt_f32_bf16_kernel<<<512, 256, 0, stream>>>(w_fc, wfc_bf, H_DIM * H_DIM);
  init_kernel<<<256, 256, 0, stream>>>(hidden, hf, hb, bar);

  // Phase 1: x projections. M=32768 -> 256 blocks of 128 rows; N=1024 -> 16x64.
  ran_proj_kernel<<<dim3(256, 16, 3), 256, 0, stream>>>(
      in_bf, wcx_bf, wix_bf, wfx_bf, b_cx, b_ix, b_fx, xc, xi, xf);

  // Phase 2: persistent scan, 32 co-resident workgroups.
  ran_scan_kernel<<<SCAN_WGS, 256, 0, stream>>>(
      wic_bf, wfc_bf, b_ic, b_fc, xc, xi, xf, hf, hb, bar, out);
}
